// LSTMCalibNet_48378511622305
// MI455X (gfx1250) — compile-verified, matrix operand order checked by reading, not run on hardware
//
#include <hip/hip_runtime.h>
#include <hip/hip_bf16.h>

// ---------------------------------------------------------------------------
// 2-layer LSTM (B=256, T=1024, H=128) + FC(128->1), PyTorch gate order i,f,g,o
// Strategy:
//   * grid = 16 WGs x 256 threads (8 wave32 waves). WG g owns batch rows
//     [16g, 16g+16) -- one WMMA M-tile. Rows are independent across WGs, and
//     layer1 rows depend only on layer0 of the SAME rows, so a single kernel
//     runs: phase0 = layer0 scan (h0 staged to ws as f16), phase1 = layer1
//     scan with combined K=256 GEMM  [h0_t | h1_{t-1}] x [W_ih1 ; W_hh1]^T,
//     fused with the final FC dot product.
//   * Per step GEMM: A = 16xK f16 (hidden state), B = Kx512 f16 (= W^T, i.e.
//     row-major W with the A-style per-lane mapping), C/D f32 via
//     v_wmma_f32_16x16x32_f16. 8 waves x 4 N-tiles = 32 N-tiles = 512 gates.
//     (Round-1 codegen confirms the compiler hoists all weight B-fragments
//     into VGPRs across the t-loop via s_set_vgpr_msb.)
//   * Elementwise gate math uses native gfx1250 transcendentals (v_tanh_f32,
//     v_exp_f32, v_rcp_f32) when the builtins exist, to shorten the serial
//     per-timestep dependency chain.
// ---------------------------------------------------------------------------

typedef __attribute__((ext_vector_type(16))) _Float16 v16h;
typedef __attribute__((ext_vector_type(8)))  _Float16 v8h;
typedef __attribute__((ext_vector_type(8)))  float    v8f;

#define BSZ   256
#define TLEN  1024
#define HD    128
#define G4    512
#define HB_LD 136   // hbuf row stride in halves (pad to dodge bank alignment)

// Per-lane fragment load for 16-bit 16x32 WMMA operands from a row-major
// matrix (works for A from row-major activations and for B == W^T loaded
// straight from row-major W, since the A/B per-lane mappings are duals).
// Lane L: row = row0 + (L&15); halves j=0..7 -> col0 + kb + j,
//         halves j=8..15 -> col0 + 16 + kb + (j-8), kb = (L>>4)*8.
static __device__ inline v16h load_frag(const _Float16* p, int ld, int row0, int col0) {
    const int lane = (int)(threadIdx.x & 31);
    const int r    = row0 + (lane & 15);
    const int kb   = (lane >> 4) << 3;
    const _Float16* q = p + r * ld + col0 + kb;
    v8h lo = *reinterpret_cast<const v8h*>(q);
    v8h hi = *reinterpret_cast<const v8h*>(q + 16);
    return __builtin_shufflevector(lo, hi, 0, 1, 2, 3, 4, 5, 6, 7,
                                           8, 9, 10, 11, 12, 13, 14, 15);
}

static __device__ inline v8f wmma_f16(v16h a, v16h b, v8f c) {
    return __builtin_amdgcn_wmma_f32_16x16x32_f16(false, a, false, b,
                                                  (short)0, c, false, false);
}

// Fast device transcendentals on the native gfx1250 units when available.
static __device__ inline float fast_rcp(float v) {
#if __has_builtin(__builtin_amdgcn_rcpf)
    return __builtin_amdgcn_rcpf(v);          // v_rcp_f32
#else
    return 1.0f / v;
#endif
}
static __device__ inline float fast_exp2(float v) {
#if __has_builtin(__builtin_amdgcn_exp2f)
    return __builtin_amdgcn_exp2f(v);         // v_exp_f32 (base-2)
#else
    return __expf(v * 0.69314718056f);
#endif
}
static __device__ inline float sigm(float v) {
    // 1 / (1 + 2^(-x * log2(e)))
    return fast_rcp(1.0f + fast_exp2(v * -1.44269504f));
}
static __device__ inline float tanh_f(float v) {
#if __has_builtin(__builtin_amdgcn_tanhf)
    return __builtin_amdgcn_tanhf(v);         // v_tanh_f32
#else
    return 2.0f * sigm(2.0f * v) - 1.0f;
#endif
}

// Convert the three recurrent/input weight matrices to f16 in workspace.
__global__ void lstm_wcvt_kernel(const float* __restrict__ Whh0,
                                 const float* __restrict__ Wih1,
                                 const float* __restrict__ Whh1,
                                 _Float16* __restrict__ w16) {
    int i = blockIdx.x * blockDim.x + threadIdx.x;     // 0 .. 3*512*128-1
    if (i < 65536)        w16[i] = (_Float16)Whh0[i];
    else if (i < 131072)  w16[i] = (_Float16)Wih1[i - 65536];
    else if (i < 196608)  w16[i] = (_Float16)Whh1[i - 131072];
}

__global__ __launch_bounds__(256) void lstm_scan_kernel(
    const float* __restrict__ x,        // [B,T,1]
    const float* __restrict__ W_ih0,    // [512,1]
    const float* __restrict__ b_ih0, const float* __restrict__ b_hh0,
    const float* __restrict__ b_ih1, const float* __restrict__ b_hh1,
    const float* __restrict__ fc_w,     // [1,128]
    const float* __restrict__ fc_b,     // [1]
    float* __restrict__ out,            // [B,T,1]
    _Float16* __restrict__ h0ws,        // [B,T,128] f16 staging
    const _Float16* __restrict__ w16)   // Whh0 | Wih1 | Whh1, each [512,128] f16
{
    __shared__ alignas(16) _Float16 hbuf[16 * HB_LD]; // current h state, f16
    __shared__ float gbuf[16 * G4];                   // gate pre-activations
    __shared__ float wxb0[G4];                        // W_ih0[:,0]
    __shared__ float bias0[G4];                       // b_ih0 + b_hh0
    __shared__ float bias1[G4];                       // b_ih1 + b_hh1
    __shared__ float fcw[HD];

    const int tid  = threadIdx.x;
    const int lane = tid & 31;
    const int wave = tid >> 5;          // 0..7
    const int b0   = blockIdx.x * 16;   // batch row base
    const int ntb  = wave * 4;          // 4 N-tiles per wave

    for (int i = tid; i < G4; i += 256) {
        wxb0[i]  = W_ih0[i];
        bias0[i] = b_ih0[i] + b_hh0[i];
        bias1[i] = b_ih1[i] + b_hh1[i];
    }
    if (tid < HD) fcw[tid] = fc_w[tid];
    for (int i = tid; i < 16 * HB_LD; i += 256) hbuf[i] = (_Float16)0.0f;

    // Elementwise mapping: row em, 8 consecutive hidden cols starting at ec.
    const int em = tid >> 4;
    const int ec = (tid & 15) * 8;
    float cst[8];
    #pragma unroll
    for (int q = 0; q < 8; ++q) cst[q] = 0.0f;
    const float fcb = fc_b[0];

    const _Float16* Whh0h = w16;
    const _Float16* Wih1h = w16 + 65536;
    const _Float16* Whh1h = w16 + 131072;

    const v8f vz = {0.f, 0.f, 0.f, 0.f, 0.f, 0.f, 0.f, 0.f};
    const int nrow = (lane >> 4) << 3;   // D-layout: lanes16-31 hold M+8
    const int ncol = lane & 15;

    __syncthreads();

    // ------------------------- PHASE 0: layer 0 ---------------------------
    for (int t = 0; t < TLEN; ++t) {
        v8f C[4] = {vz, vz, vz, vz};
        #pragma unroll
        for (int kt = 0; kt < 4; ++kt) {
            v16h A = load_frag(hbuf, HB_LD, 0, kt * 32);
            #pragma unroll
            for (int j = 0; j < 4; ++j) {
                v16h Bf = load_frag(Whh0h, HD, (ntb + j) * 16, kt * 32);
                C[j] = wmma_f16(A, Bf, C[j]);
            }
        }
        #pragma unroll
        for (int j = 0; j < 4; ++j) {
            const int nc = (ntb + j) * 16 + ncol;
            #pragma unroll
            for (int r = 0; r < 8; ++r) gbuf[(nrow + r) * G4 + nc] = C[j][r];
        }
        __syncthreads();

        const float xv = x[(b0 + em) * TLEN + t];
        v8h hv;
        float* gb = &gbuf[em * G4];
        #pragma unroll
        for (int q = 0; q < 8; ++q) {
            const int hc = ec + q;
            float gi = gb[hc]       + xv * wxb0[hc]       + bias0[hc];
            float gf = gb[128 + hc] + xv * wxb0[128 + hc] + bias0[128 + hc];
            float gg = gb[256 + hc] + xv * wxb0[256 + hc] + bias0[256 + hc];
            float go = gb[384 + hc] + xv * wxb0[384 + hc] + bias0[384 + hc];
            float c  = sigm(gf) * cst[q] + sigm(gi) * tanh_f(gg);
            cst[q]   = c;
            hv[q]    = (_Float16)(sigm(go) * tanh_f(c));
        }
        *reinterpret_cast<v8h*>(&hbuf[em * HB_LD + ec]) = hv;
        *reinterpret_cast<v8h*>(&h0ws[((size_t)(b0 + em) * TLEN + t) * HD + ec]) = hv;
        __syncthreads();
    }

    // reset state between layers; make h0ws visible across waves
    __threadfence();
    for (int i = tid; i < 16 * HB_LD; i += 256) hbuf[i] = (_Float16)0.0f;
    #pragma unroll
    for (int q = 0; q < 8; ++q) cst[q] = 0.0f;
    __syncthreads();

    // ------------------ PHASE 1: layer 1 + FC, K = 256 --------------------
    for (int t = 0; t < TLEN; ++t) {
        const _Float16* h0base = h0ws + ((size_t)b0 * TLEN + t) * HD;

        // Prefetch next timestep's h0 tile (L2-resident -> near cache).
        if (t + 1 < TLEN) {
            const _Float16* nb = h0base + HD;                 // t+1 same rows
            __builtin_prefetch(nb + (size_t)(lane & 15) * (TLEN * HD)
                                  + (lane >> 4) * 64, 0, 3);
        }

        v8f C[4] = {vz, vz, vz, vz};
        #pragma unroll
        for (int kt = 0; kt < 8; ++kt) {
            v16h A = (kt < 4) ? load_frag(h0base, TLEN * HD, 0, kt * 32)
                              : load_frag(hbuf, HB_LD, 0, (kt - 4) * 32);
            const _Float16* Wp = (kt < 4) ? Wih1h : Whh1h;
            const int wc = (kt & 3) * 32;
            #pragma unroll
            for (int j = 0; j < 4; ++j) {
                v16h Bf = load_frag(Wp, HD, (ntb + j) * 16, wc);
                C[j] = wmma_f16(A, Bf, C[j]);
            }
        }
        #pragma unroll
        for (int j = 0; j < 4; ++j) {
            const int nc = (ntb + j) * 16 + ncol;
            #pragma unroll
            for (int r = 0; r < 8; ++r) gbuf[(nrow + r) * G4 + nc] = C[j][r];
        }
        __syncthreads();

        v8h hv;
        float dot = 0.0f;
        float* gb = &gbuf[em * G4];
        #pragma unroll
        for (int q = 0; q < 8; ++q) {
            const int hc = ec + q;
            float gi = gb[hc]       + bias1[hc];
            float gf = gb[128 + hc] + bias1[128 + hc];
            float gg = gb[256 + hc] + bias1[256 + hc];
            float go = gb[384 + hc] + bias1[384 + hc];
            float c  = sigm(gf) * cst[q] + sigm(gi) * tanh_f(gg);
            cst[q]   = c;
            float h  = sigm(go) * tanh_f(c);
            hv[q]    = (_Float16)h;
            dot     += h * fcw[hc];
        }
        *reinterpret_cast<v8h*>(&hbuf[em * HB_LD + ec]) = hv;
        // reduce the 16 threads of row em (they are a half-wave)
        #pragma unroll
        for (int off = 1; off < 16; off <<= 1) dot += __shfl_xor(dot, off, 32);
        if ((tid & 15) == 0) out[(b0 + em) * TLEN + t] = dot + fcb;
        __syncthreads();
    }
}

extern "C" void kernel_launch(void* const* d_in, const int* in_sizes, int n_in,
                              void* d_out, int out_size, void* d_ws, size_t ws_size,
                              hipStream_t stream) {
    const float* x     = (const float*)d_in[0];
    const float* W_ih0 = (const float*)d_in[1];
    const float* W_hh0 = (const float*)d_in[2];
    const float* b_ih0 = (const float*)d_in[3];
    const float* b_hh0 = (const float*)d_in[4];
    const float* W_ih1 = (const float*)d_in[5];
    const float* W_hh1 = (const float*)d_in[6];
    const float* b_ih1 = (const float*)d_in[7];
    const float* b_hh1 = (const float*)d_in[8];
    const float* fc_w  = (const float*)d_in[9];
    const float* fc_b  = (const float*)d_in[10];
    float* out = (float*)d_out;

    // ws layout: [0, 64MB) h0 staging f16, then 384KB of f16 weights
    _Float16* h0ws = (_Float16*)d_ws;
    _Float16* w16  = h0ws + (size_t)BSZ * TLEN * HD;

    lstm_wcvt_kernel<<<768, 256, 0, stream>>>(W_hh0, W_ih1, W_hh1, w16);
    lstm_scan_kernel<<<BSZ / 16, 256, 0, stream>>>(
        x, W_ih0, b_ih0, b_hh0, b_ih1, b_hh1, fc_w, fc_b, out, h0ws, w16);
}
